// SelfAttention_22127671509167
// MI455X (gfx1250) — matrix operand order repaired and checked by reading, not compiled, not run on hardware
//
#include <hip/hip_runtime.h>

typedef unsigned int   u32;
typedef unsigned short u16;

typedef __attribute__((ext_vector_type(16))) __bf16 v16bf;
typedef __attribute__((ext_vector_type(8)))  __bf16 v8bf;
typedef __attribute__((ext_vector_type(8)))  float  v8f;

union AFrag { v16bf v; v8bf h[2]; };

#define BATCH 4
#define SEQ   4096
#define DIN   512
#define HID   256
#define NTOT  1280   // 512 (Q) + 512 (K) + 256 (V)

#ifndef __has_builtin
#define __has_builtin(x) 0
#endif
#if __has_builtin(__builtin_amdgcn_tensor_load_to_lds) && \
    __has_builtin(__builtin_amdgcn_s_wait_tensorcnt)
#define HAVE_TDM 1
#else
#define HAVE_TDM 0
#endif

// fp32 -> bf16, round-to-nearest-even
__device__ __forceinline__ u16 f2bf(float f) {
  u32 u = __float_as_uint(f);
  u += 0x7FFFu + ((u >> 16) & 1u);
  return (u16)(u >> 16);
}
__device__ __forceinline__ u32 pk2(float a, float b) {
  return (u32)f2bf(a) | ((u32)f2bf(b) << 16);
}
__device__ __forceinline__ v8f wmma_bf16(const AFrag& a, const v16bf& b, v8f c) {
  // D(16x16 f32) = A(16x32 bf16) * B(32x16 bf16) + C
  return __builtin_amdgcn_wmma_f32_16x16x32_bf16(false, a.v, false, b, (short)0, c,
                                                 false, false);
}

#if HAVE_TDM
typedef __attribute__((ext_vector_type(4))) unsigned int tdm_g0_t;
typedef __attribute__((ext_vector_type(8))) int          tdm_g1_t;
typedef __attribute__((ext_vector_type(4))) int          tdm_g23_t;

// Issue one 2D TDM tile load (bf16 elements, data_size=2B):
//   global tile [tile1 rows x tile0 elems], row stride = stride0 elems,
//   packed contiguously into LDS at byte offset ldsAddr.
// D# bit layout per CDNA5 ISA ch.8 (group0: count/lds/global/type=2,
// group1: data_size=1(2B), dims/strides/tile dims). Groups 2/3 zero (2D).
// 6-arg builtin on this toolchain: (g0 v4u, g1 v8i, g2 v4i, g3 v4i, v8i, cpol).
__device__ __forceinline__ void tdm_load_2d(unsigned ldsAddr, const void* gptr,
                                            unsigned dim0, unsigned stride0,
                                            unsigned tile0, unsigned tile1) {
  unsigned long long ga = (unsigned long long)(size_t)gptr;
  const unsigned dim1 = 1u << 20;  // generous row bound (OOB rows read as 0)
  tdm_g0_t g0 = { 1u,                                   // count=1 valid descriptor
                  ldsAddr,                              // lds_addr [63:32]
                  (unsigned)ga,                         // global_addr lo
                  (unsigned)((ga >> 32) & 0x01FFFFFFu)  // global_addr hi (57b)
                      | (2u << 30) };                   // type=2 ("image")
  tdm_g1_t g1 = { (int)0x00010000u,                     // wg_mask=0, data_size=1 (2B)
                  (int)((dim0 & 0xFFFFu) << 16),        // tensor_dim0 lo16
                  (int)((dim0 >> 16) | ((dim1 & 0xFFFFu) << 16)), // dim0 hi / dim1 lo
                  (int)((dim1 >> 16) | (tile0 << 16)),  // dim1 hi / tile_dim0
                  (int)(tile1 & 0xFFFFu),               // tile_dim1 (tile_dim2=0)
                  (int)stride0,                         // tensor_dim0_stride lo32
                  0,                                    // stride0 hi / dim1_stride lo
                  0 };
  tdm_g23_t gz = { 0, 0, 0, 0 };
  tdm_g1_t  gx = { 0, 0, 0, 0, 0, 0, 0, 0 };
  __builtin_amdgcn_tensor_load_to_lds(g0, g1, gz, gz, gx, 0);
}
#endif

// ---------------------------------------------------------------------------
// Kernel 1: fused QKV projection.  C[64x128] per block, K-chunks of 32.
// x and W converted fp32->bf16 while staging into LDS. Bs kept column-major
// so each B fragment is one contiguous 32B LDS load (32x16 bf16 B layout).
// Q,K written row-major bf16; V written TRANSPOSED (Vt[b][d][seq]) with
// packed uint4 stores so the attention kernel streams V rows contiguously.
// ---------------------------------------------------------------------------
__global__ __launch_bounds__(256)
void qkv_proj_kernel(const float* __restrict__ x,
                     const float* __restrict__ Wq, const float* __restrict__ bq,
                     const float* __restrict__ Wk, const float* __restrict__ bk,
                     const float* __restrict__ Wv, const float* __restrict__ bv,
                     u16* __restrict__ Qb, u16* __restrict__ Kb, u16* __restrict__ Vt)
{
  __shared__ u16 As[64 * 32];    // [m][k]
  __shared__ u16 Bs[128 * 32];   // [n][k]  (column-major B)

  const int tid  = threadIdx.x;
  const int lane = tid & 31;
  const int wv   = tid >> 5;     // wave 0..7, grid 2(M) x 4(N)
  const int ln   = lane & 15;
  const int hh   = lane >> 4;

  const int b  = blockIdx.z;
  const int m0 = blockIdx.y * 64;
  const int n0 = blockIdx.x * 128;      // global col in concatenated [Q|K|V]

  const float* Wseg; const float* bseg; u16* dst; int segcols; int c0;
  if (n0 < 512)       { Wseg = Wq; bseg = bq; dst = Qb; segcols = 512; c0 = n0;        }
  else if (n0 < 1024) { Wseg = Wk; bseg = bk; dst = Kb; segcols = 512; c0 = n0 - 512;  }
  else                { Wseg = Wv; bseg = bv; dst = Vt; segcols = 256; c0 = n0 - 1024; }
  const bool isV = (n0 >= 1024);

  v8f acc[2][2] = {};

  const int arow = tid >> 2;          // 0..63
  const int ako  = (tid & 3) * 8;     // 0,8,16,24
  const int bkk  = tid >> 3;          // 0..31
  const int bc0  = (tid & 7) * 16;    // 0..112

  for (int k0 = 0; k0 < DIN; k0 += 32) {
    // stage A (x) tile, fp32 -> bf16
    const float* ax = x + ((size_t)(b * SEQ + m0 + arow) * DIN + k0 + ako);
    float4 f0 = *(const float4*)ax;
    float4 f1 = *(const float4*)(ax + 4);
    uint4 pa;
    pa.x = pk2(f0.x, f0.y); pa.y = pk2(f0.z, f0.w);
    pa.z = pk2(f1.x, f1.y); pa.w = pk2(f1.z, f1.w);
    *(uint4*)&As[arow * 32 + ako] = pa;
    // stage B (W) tile transposed to [n][k], fp32 -> bf16
    const float* wx = Wseg + ((size_t)(k0 + bkk) * segcols + c0 + bc0);
#pragma unroll
    for (int i = 0; i < 4; ++i) {
      float4 w = *(const float4*)(wx + i * 4);
      Bs[(bc0 + i*4 + 0) * 32 + bkk] = f2bf(w.x);
      Bs[(bc0 + i*4 + 1) * 32 + bkk] = f2bf(w.y);
      Bs[(bc0 + i*4 + 2) * 32 + bkk] = f2bf(w.z);
      Bs[(bc0 + i*4 + 3) * 32 + bkk] = f2bf(w.w);
    }
    __syncthreads();
#pragma unroll
    for (int mt = 0; mt < 2; ++mt) {
      AFrag a;
      const u16* ap = &As[(32 * (wv >> 2) + 16 * mt + ln) * 32 + hh * 16];
      a.h[0] = *(const v8bf*)ap;
      a.h[1] = *(const v8bf*)(ap + 8);
#pragma unroll
      for (int nt = 0; nt < 2; ++nt) {
        const u16* bp = &Bs[(32 * (wv & 3) + 16 * nt + ln) * 32 + hh * 16];
        v16bf bb = *(const v16bf*)bp;
        acc[mt][nt] = wmma_bf16(a, bb, acc[mt][nt]);
      }
    }
    __syncthreads();
  }

  // epilogue: + bias, store bf16
#pragma unroll
  for (int mt = 0; mt < 2; ++mt)
#pragma unroll
    for (int nt = 0; nt < 2; ++nt) {
      int coll = 32 * (wv & 3) + 16 * nt + ln;
      float bias = bseg[c0 + coll];
      if (!isV) {
#pragma unroll
        for (int j = 0; j < 8; ++j) {
          int row = m0 + 32 * (wv >> 2) + 16 * mt + j + 8 * hh;
          dst[(size_t)(b * SEQ + row) * segcols + c0 + coll] =
              f2bf(acc[mt][nt][j] + bias);
        }
      } else {
        // transposed store: rows j are consecutive seq positions -> uint4
        int rowbase = m0 + 32 * (wv >> 2) + 16 * mt + 8 * hh;
        uint4 pv;
        pv.x = pk2(acc[mt][nt][0] + bias, acc[mt][nt][1] + bias);
        pv.y = pk2(acc[mt][nt][2] + bias, acc[mt][nt][3] + bias);
        pv.z = pk2(acc[mt][nt][4] + bias, acc[mt][nt][5] + bias);
        pv.w = pk2(acc[mt][nt][6] + bias, acc[mt][nt][7] + bias);
        *(uint4*)&dst[((size_t)(b * HID) + c0 + coll) * SEQ + rowbase] = pv;
      }
    }
}

// ---------------------------------------------------------------------------
// Kernel 2: flash attention.  32 query rows per block, 64-key tiles,
// head-dim streamed through a DOUBLE-BUFFERED 2x(64x64) bf16 LDS buffer.
// With TDM: wave 0 keeps one DMA transfer in flight behind the WMMA work
// (issue next chunk, s_wait_tensorcnt 1 for the chunk about to be consumed —
// TDM transfers from one wave complete in order).  Fallback: vector copies.
// ---------------------------------------------------------------------------
__global__ __launch_bounds__(256)
void attn_kernel(const u16* __restrict__ Qb, const u16* __restrict__ Kb,
                 const u16* __restrict__ Vt, float* __restrict__ out)
{
  __shared__ u16   Qs[32 * 512];   // query tile, bf16        (32 KB)
  __shared__ u16   KV[2][64 * 64]; // K / V^T chunks, 2 bufs  (16 KB)
  __shared__ float Ssh[32 * 64];   // score tile, fp32        ( 8 KB)
  __shared__ u16   Ps[32 * 64];    // softmax probs, bf16     ( 4 KB)
  __shared__ float mrow[32], lrow[32], arow[32];
  __shared__ float red[32 * 8];

  const int tid  = threadIdx.x;
  const int lane = tid & 31, wv = tid >> 5;
  const int ln   = lane & 15, hh = lane >> 4;
  const int Mt   = wv >> 2,   wn = wv & 3;   // wave grid: 2(M) x 4(N)

  const int b  = blockIdx.y;
  const int q0 = blockIdx.x * 32;

#if HAVE_TDM
  // generic LDS pointers carry the LDS byte offset in their low 32 bits
  const unsigned kvA0 = (unsigned)(size_t)(const void*)&KV[0][0];
  const unsigned kvA1 = (unsigned)(size_t)(const void*)&KV[1][0];
  const unsigned qsAddr = (unsigned)(size_t)(const void*)&Qs[0];
  if (wv == 0) {
    tdm_load_2d(qsAddr, Qb + (size_t)(b * SEQ + q0) * 512, 512, 512, 512, 32);
    __builtin_amdgcn_s_wait_tensorcnt(0);
  }
#else
  { // load Q tile (already bf16)
    int row = tid >> 3, seg = (tid & 7) * 64;
    const uint4* src = (const uint4*)(Qb + ((size_t)(b * SEQ + q0 + row) * 512 + seg));
    uint4* dq = (uint4*)&Qs[row * 512 + seg];
#pragma unroll
    for (int i = 0; i < 8; ++i) dq[i] = src[i];
  }
#endif
  if (tid < 32) { mrow[tid] = -1e30f; lrow[tid] = 0.f; arow[tid] = 0.f; }
  __syncthreads();

  v8f acc[4] = {};                         // O tile: 32 rows x 256 cols

  const int srow = tid >> 3, sp = tid & 7; // softmax: 8 threads / row
  const int kkey = tid >> 2, kdo = (tid & 3) * 16;

  for (int it = 0; it < 64; ++it) {
    const int key0 = it * 64;
    const u16* Krow = Kb + (size_t)(b * SEQ + key0) * 512;

    // ---- S = Q . K^T over 512 dims, 64-wide double-buffered chunks -------
    v8f s = {};
#if HAVE_TDM
    __syncthreads();                       // prior readers of KV are done
    if (wv == 0)
      tdm_load_2d(kvA0, Krow, 512, 512, 64, 64);
#endif
#pragma unroll 1
    for (int kc = 0; kc < 8; ++kc) {
      __syncthreads();                     // readers of the other buffer done
#if HAVE_TDM
      if (wv == 0) {
        if (kc < 7) {
          tdm_load_2d((kc & 1) ? kvA0 : kvA1, Krow + (kc + 1) * 64,
                      512, 512, 64, 64);
          __builtin_amdgcn_s_wait_tensorcnt(1);  // chunk kc has landed
        } else {
          __builtin_amdgcn_s_wait_tensorcnt(0);
        }
      }
#else
      {
        const uint4* ksrc =
            (const uint4*)(Krow + (size_t)kkey * 512 + kc * 64 + kdo);
        uint4* kdst = (uint4*)&KV[kc & 1][kkey * 64 + kdo];
        kdst[0] = ksrc[0]; kdst[1] = ksrc[1];
      }
#endif
      if (it + 1 < 64)
        __builtin_prefetch(
            (const void*)(Kb + ((size_t)(b * SEQ + key0 + 64 + kkey) * 512 + kc * 64 + kdo)),
            0, 0);
      __syncthreads();
      const u16* kvb = &KV[kc & 1][0];
#pragma unroll
      for (int sub = 0; sub < 2; ++sub) {
        AFrag a;
        const u16* ap = &Qs[(16 * Mt + ln) * 512 + kc * 64 + sub * 32 + hh * 16];
        a.h[0] = *(const v8bf*)ap; a.h[1] = *(const v8bf*)(ap + 8);
        const u16* bp = &kvb[(16 * wn + ln) * 64 + sub * 32 + hh * 16];
        v16bf bb = *(const v16bf*)bp;
        s = wmma_bf16(a, bb, s);
      }
    }
    // scale = 1/sqrt(HIDDEN) = 1/16 ; spill tile to LDS
#pragma unroll
    for (int j = 0; j < 8; ++j)
      Ssh[(16 * Mt + j + 8 * hh) * 64 + 16 * wn + ln] = s[j] * 0.0625f;
    __syncthreads();

    // ---- online softmax --------------------------------------------------
    float tm = -1e30f;
#pragma unroll
    for (int c = 0; c < 8; ++c) tm = fmaxf(tm, Ssh[srow * 64 + sp * 8 + c]);
    red[srow * 8 + sp] = tm;
    __syncthreads();
    if (sp == 0) {
      float mo = mrow[srow], mx = mo;
#pragma unroll
      for (int qv = 0; qv < 8; ++qv) mx = fmaxf(mx, red[srow * 8 + qv]);
      mrow[srow] = mx;
      arow[srow] = __expf(mo - mx);
    }
    __syncthreads();
    float mr = mrow[srow], psum = 0.f;
#pragma unroll
    for (int c = 0; c < 8; ++c) {
      int idx = srow * 64 + sp * 8 + c;
      float e = __expf(Ssh[idx] - mr);
      Ps[idx] = f2bf(e);
      psum += e;
    }
    red[srow * 8 + sp] = psum;
    __syncthreads();
    if (sp == 0) {
      float sacc = 0.f;
#pragma unroll
      for (int qv = 0; qv < 8; ++qv) sacc += red[srow * 8 + qv];
      lrow[srow] = lrow[srow] * arow[srow] + sacc;
    }
    __syncthreads();

    // rescale O accumulators by per-row alpha
    float al[8];
#pragma unroll
    for (int j = 0; j < 8; ++j) al[j] = arow[16 * Mt + j + 8 * hh];
#pragma unroll
    for (int c = 0; c < 4; ++c)
#pragma unroll
      for (int j = 0; j < 8; ++j) acc[c][j] *= al[j];

    // P fragments (reused across the 4 V chunks)
    AFrag pa[2];
#pragma unroll
    for (int kk = 0; kk < 2; ++kk) {
      const u16* pp = &Ps[(16 * Mt + ln) * 64 + kk * 32 + hh * 16];
      pa[kk].h[0] = *(const v8bf*)pp; pa[kk].h[1] = *(const v8bf*)(pp + 8);
    }

    // ---- O += P . V, streaming V^T in 64-dim double-buffered chunks ------
    const u16* Vrow = Vt + (size_t)(b * HID) * SEQ + key0;
#if HAVE_TDM
    __syncthreads();                       // S-phase readers of KV done
    if (wv == 0)
      tdm_load_2d(kvA0, Vrow, SEQ, SEQ, 64, 64);
#endif
#pragma unroll 1
    for (int c = 0; c < 4; ++c) {
      __syncthreads();
#if HAVE_TDM
      if (wv == 0) {
        if (c < 3) {
          tdm_load_2d((c & 1) ? kvA0 : kvA1, Vrow + (size_t)((c + 1) * 64) * SEQ,
                      SEQ, SEQ, 64, 64);
          __builtin_amdgcn_s_wait_tensorcnt(1);  // chunk c has landed
        } else {
          __builtin_amdgcn_s_wait_tensorcnt(0);
        }
      }
#else
      {
        const int dl = tid >> 2, ko = (tid & 3) * 16;
        const uint4* vsrc =
            (const uint4*)(Vrow + (size_t)(c * 64 + dl) * SEQ + ko);
        uint4* vdst = (uint4*)&KV[c & 1][dl * 64 + ko];
        vdst[0] = vsrc[0]; vdst[1] = vsrc[1];
      }
#endif
      if (it + 1 < 64)
        __builtin_prefetch(
            (const void*)(Vt + ((size_t)(b * HID + c * 64 + (tid >> 2)) * SEQ + key0 + 64)),
            0, 0);
      __syncthreads();
      const u16* kvb = &KV[c & 1][0];
#pragma unroll
      for (int sub = 0; sub < 2; ++sub) {
        const u16* bp = &kvb[(16 * wn + ln) * 64 + sub * 32 + hh * 16];
        v16bf bb = *(const v16bf*)bp;
        acc[c] = wmma_bf16(pa[sub], bb, acc[c]);
      }
    }
  }

  __syncthreads();
  // normalize and store fp32 output [B,S,256]
#pragma unroll
  for (int j = 0; j < 8; ++j) {
    int row = 16 * Mt + j + 8 * hh;
    float inv = 1.0f / lrow[row];
#pragma unroll
    for (int c = 0; c < 4; ++c)
      out[(size_t)(b * SEQ + q0 + row) * 256 + 64 * c + 16 * wn + ln] = acc[c][j] * inv;
  }
}

// ---------------------------------------------------------------------------
extern "C" void kernel_launch(void* const* d_in, const int* in_sizes, int n_in,
                              void* d_out, int out_size, void* d_ws, size_t ws_size,
                              hipStream_t stream) {
  (void)in_sizes; (void)n_in; (void)out_size; (void)ws_size;
  const float* x  = (const float*)d_in[0];
  const float* Wq = (const float*)d_in[1];
  const float* bq = (const float*)d_in[2];
  const float* Wk = (const float*)d_in[3];
  const float* bk = (const float*)d_in[4];
  const float* Wv = (const float*)d_in[5];
  const float* bv = (const float*)d_in[6];
  float* out = (float*)d_out;

  // workspace: bf16 Q (16MB) | bf16 K (16MB) | bf16 V^T (8MB)
  u16* Qb = (u16*)d_ws;
  u16* Kb = Qb + (size_t)BATCH * SEQ * 512;
  u16* Vt = Kb + (size_t)BATCH * SEQ * 512;

  dim3 gB(NTOT / 128, SEQ / 64, BATCH), tB(256);
  qkv_proj_kernel<<<gB, tB, 0, stream>>>(x, Wq, bq, Wk, bk, Wv, bv, Qb, Kb, Vt);

  dim3 gC(SEQ / 32, BATCH), tC(256);
  attn_kernel<<<gC, tC, 0, stream>>>(Qb, Kb, Vt, out);
}
